// PointNet2A_6682969112987
// MI455X (gfx1250) — matrix operand order repaired and checked by reading, not compiled, not run on hardware
//
#include <hip/hip_runtime.h>
#include <hip/hip_bf16.h>

typedef __attribute__((ext_vector_type(16))) _Float16 v16h;
typedef __attribute__((ext_vector_type(8)))  float    v8f;

#define FPS_T 256

// ---------------- slice xyz[:, :, :3] ----------------
__global__ void k_slice_xyz(const float* __restrict__ x, float* __restrict__ o, int BN) {
    int t = blockIdx.x * blockDim.x + threadIdx.x;
    if (t >= BN) return;
    o[(size_t)t * 3 + 0] = x[(size_t)t * 6 + 0];
    o[(size_t)t * 3 + 1] = x[(size_t)t * 6 + 1];
    o[(size_t)t * 3 + 2] = x[(size_t)t * 6 + 2];
}

// ---------------- farthest point sampling (one block per batch) ----------------
__global__ void k_fps(const float* __restrict__ xyz, int* __restrict__ fidx, int N, int npoint) {
    __shared__ float sDist[8192];
    __shared__ float sVal[FPS_T];
    __shared__ int   sIdx[FPS_T];
    __shared__ int   sFar;
    __shared__ float sC[3];
    int b = blockIdx.x, t = threadIdx.x;
    const float* X = xyz + (size_t)b * N * 3;
    for (int i = t; i < N; i += FPS_T) sDist[i] = 1e10f;
    if (t == 0) sFar = 0;
    __syncthreads();
    for (int it = 0; it < npoint; ++it) {
        int far = sFar;
        if (t == 0) {
            fidx[b * npoint + it] = far;
            sC[0] = X[far * 3 + 0]; sC[1] = X[far * 3 + 1]; sC[2] = X[far * 3 + 2];
        }
        __syncthreads();
        float cx = sC[0], cy = sC[1], cz = sC[2];
        float bv = -1.0f; int bi = N;
        for (int i = t; i < N; i += FPS_T) {
            float dx = X[i * 3 + 0] - cx, dy = X[i * 3 + 1] - cy, dz = X[i * 3 + 2] - cz;
            float d = dx * dx + dy * dy + dz * dz;
            float nd = fminf(sDist[i], d);
            sDist[i] = nd;
            if (nd > bv) { bv = nd; bi = i; }
        }
        sVal[t] = bv; sIdx[t] = bi;
        __syncthreads();
        for (int s2 = FPS_T / 2; s2 > 0; s2 >>= 1) {
            if (t < s2) {
                float ov = sVal[t + s2]; int oi = sIdx[t + s2];
                if (ov > sVal[t] || (ov == sVal[t] && oi < sIdx[t])) { sVal[t] = ov; sIdx[t] = oi; }
            }
            __syncthreads();
        }
        if (t == 0) sFar = sIdx[0];
        __syncthreads();
    }
}

// ---------------- gather new_xyz = xyz[fidx] ----------------
__global__ void k_gather_xyz(const float* __restrict__ xyz, const int* __restrict__ fidx,
                             float* __restrict__ o, int B, int N, int S) {
    int t = blockIdx.x * blockDim.x + threadIdx.x;
    if (t >= B * S) return;
    int b = t / S;
    int idx = fidx[t];
    const float* p = xyz + ((size_t)b * N + idx) * 3;
    o[(size_t)t * 3 + 0] = p[0];
    o[(size_t)t * 3 + 1] = p[1];
    o[(size_t)t * 3 + 2] = p[2];
}

// ---------------- ball query: first `ns` points (in index order) with d^2 <= r^2 ----------------
__global__ void k_ballquery(const float* __restrict__ xyz, const float* __restrict__ nxyz,
                            int* __restrict__ gidx, int B, int N, int S, float r2, int ns) {
    int t = blockIdx.x * blockDim.x + threadIdx.x;
    if (t >= B * S) return;
    int b = t / S;
    const float* X = xyz + (size_t)b * N * 3;
    float qx = nxyz[(size_t)t * 3], qy = nxyz[(size_t)t * 3 + 1], qz = nxyz[(size_t)t * 3 + 2];
    int* o = gidx + (size_t)t * ns;
    int cnt = 0;
    for (int n = 0; n < N && cnt < ns; ++n) {
        float dx = X[n * 3] - qx, dy = X[n * 3 + 1] - qy, dz = X[n * 3 + 2] - qz;
        if (dx * dx + dy * dy + dz * dz <= r2) o[cnt++] = n;
    }
    int f = (cnt > 0) ? o[0] : 0;
    for (; cnt < ns; ++cnt) o[cnt] = f;
}

// ---------------- group: g = [xyz[gidx]-new_xyz, points[gidx], 0-pad to Kp] ----------------
__global__ void k_group(const float* __restrict__ xyz, const float* __restrict__ pts,
                        const float* __restrict__ nxyz, const int* __restrict__ gidx,
                        float* __restrict__ g, int B, int N, int S, int ns, int C, int Kp) {
    int t = blockIdx.x * blockDim.x + threadIdx.x;
    if (t >= B * S * ns) return;
    int bs = t / ns;
    int b = bs / S;
    int idx = gidx[t];
    const float* P  = pts + ((size_t)b * N + idx) * C;
    const float* Xp = xyz + ((size_t)b * N + idx) * 3;
    const float* q  = nxyz + (size_t)bs * 3;
    float* o = g + (size_t)t * Kp;
    o[0] = Xp[0] - q[0]; o[1] = Xp[1] - q[1]; o[2] = Xp[2] - q[2];
    for (int c = 0; c < C; ++c) o[3 + c] = P[c];
    for (int c = 3 + C; c < Kp; ++c) o[c] = 0.0f;
}

// ---------------- weight pre-pack into B-fragment layout (f16, zero-padded K) ----------------
// out layout: [Kp/32][N/16][lane 0..31][e 0..15], element = W[k][n] with
//   half = lane>>4, lr = lane&15, k = ks*32 + 16*half + e, n = nt*16 + lr
__global__ void k_packW(const float* __restrict__ W, _Float16* __restrict__ o, int K, int Kp, int N) {
    int t = blockIdx.x * blockDim.x + threadIdx.x;
    int nt16 = N >> 4;
    int total = (Kp >> 5) * nt16 * 512;
    if (t >= total) return;
    int e    = t & 15;
    int lane = (t >> 4) & 31;
    int nt   = (t >> 9) % nt16;
    int ks   = t / (512 * nt16);
    int half = lane >> 4, lr = lane & 15;
    int k = ks * 32 + 16 * half + e;
    int n = nt * 16 + lr;
    o[t] = (_Float16)((k < K) ? W[(size_t)k * N + n] : 0.0f);
}

// ---------------- WMMA GEMM + bias + BN-fold + ReLU ----------------
// A: M x Kp (f32, zero-padded), Wp: packed f16 fragments, out: M x N (f32).
// Block = 4 waves; each wave owns a 16(M) x 32(N) output (2 WMMA tiles, A reused).
// M % 64 == 0, N % 32 == 0, Kp % 32 == 0  -> no guards anywhere.
__global__ __launch_bounds__(128) void k_gemm(const float* __restrict__ A,
                                              const _Float16* __restrict__ Wp,
                                              const float* __restrict__ bias,
                                              const float* __restrict__ gamma,
                                              const float* __restrict__ beta,
                                              float* __restrict__ out,
                                              int M, int Kp, int N) {
    int lane = threadIdx.x & 31;
    int wave = threadIdx.x >> 5;
    int m0 = (blockIdx.x * 4 + wave) * 16;
    int n0 = blockIdx.y * 32;
    int half = lane >> 4;
    int lr   = lane & 15;
    int arow = m0 + lr;
    int ntiles = N >> 4;
    int nt0 = n0 >> 4;

    const float* Arow = A + (size_t)arow * Kp;
    v8f acc0 = {}, acc1 = {};

    for (int k0 = 0, ks = 0; k0 < Kp; k0 += 32, ++ks) {
        // A fragment: one contiguous 16-float run per lane, k = k0 + 16*half + e (b128 loads, no guards)
        const float4* s0 = (const float4*)(Arow + k0 + 16 * half);
        const float4* s1 = (const float4*)(Arow + k0 + 16 * half + 8);
        float4 f0 = s0[0], f1 = s0[1];
        float4 f2 = s1[0], f3 = s1[1];
        v16h a;
        a[0] = (_Float16)f0.x; a[1] = (_Float16)f0.y; a[2] = (_Float16)f0.z; a[3] = (_Float16)f0.w;
        a[4] = (_Float16)f1.x; a[5] = (_Float16)f1.y; a[6] = (_Float16)f1.z; a[7] = (_Float16)f1.w;
        a[8]  = (_Float16)f2.x; a[9]  = (_Float16)f2.y; a[10] = (_Float16)f2.z; a[11] = (_Float16)f2.w;
        a[12] = (_Float16)f3.x; a[13] = (_Float16)f3.y; a[14] = (_Float16)f3.z; a[15] = (_Float16)f3.w;
        // B fragments: one contiguous 32B load per lane per tile
        v16h b0 = *(const v16h*)(Wp + ((size_t)(ks * ntiles + nt0)     * 512 + lane * 16));
        v16h b1 = *(const v16h*)(Wp + ((size_t)(ks * ntiles + nt0 + 1) * 512 + lane * 16));
        acc0 = __builtin_amdgcn_wmma_f32_16x16x32_f16(false, a, false, b0, (short)0, acc0, false, false);
        acc1 = __builtin_amdgcn_wmma_f32_16x16x32_f16(false, a, false, b1, (short)0, acc1, false, false);
    }

    const float inv = rsqrtf(1.0f + 1e-5f);
    int c0 = n0 + lr, c1 = n0 + 16 + lr;
    float bs0 = bias[c0], sc0 = gamma[c0] * inv, bt0 = beta[c0];
    float bs1 = bias[c1], sc1 = gamma[c1] * inv, bt1 = beta[c1];
    #pragma unroll
    for (int r = 0; r < 8; ++r) {
        int m = m0 + r + 8 * half;
        float v0 = fmaxf((acc0[r] + bs0) * sc0 + bt0, 0.0f);
        float v1 = fmaxf((acc1[r] + bs1) * sc1 + bt1, 0.0f);
        out[(size_t)m * N + c0] = v0;
        out[(size_t)m * N + c1] = v1;
    }
}

// ---------------- max pool over neighborhood dim ----------------
__global__ void k_maxpool(const float* __restrict__ f, float* __restrict__ o,
                          int rows, int ns, int C) {
    int t = blockIdx.x * blockDim.x + threadIdx.x;
    if (t >= rows * C) return;
    int bs = t / C, c = t % C;
    const float* p = f + (size_t)bs * ns * C + c;
    float m = p[0];
    for (int j = 1; j < ns; ++j) m = fmaxf(m, p[(size_t)j * C]);
    o[t] = m;
}

// ---------------- FP: 3-NN inverse-distance interpolation + concat ----------------
__global__ void k_fp_interp(const float* __restrict__ xyz1, const float* __restrict__ xyz2,
                            const float* __restrict__ p1, int C1,
                            const float* __restrict__ p2, int C2,
                            float* __restrict__ out, int B, int Nn, int S) {
    int t = blockIdx.x * blockDim.x + threadIdx.x;
    if (t >= B * Nn) return;
    int b = t / Nn;
    float qx = xyz1[(size_t)t * 3], qy = xyz1[(size_t)t * 3 + 1], qz = xyz1[(size_t)t * 3 + 2];
    const float* X2 = xyz2 + (size_t)b * S * 3;
    float d0 = 1e30f, d1 = 1e30f, d2 = 1e30f;
    int   i0 = 0,     i1 = 0,     i2 = 0;
    for (int s = 0; s < S; ++s) {
        float dx = X2[s * 3] - qx, dy = X2[s * 3 + 1] - qy, dz = X2[s * 3 + 2] - qz;
        float d = dx * dx + dy * dy + dz * dz;
        if (d < d0)      { d2 = d1; i2 = i1; d1 = d0; i1 = i0; d0 = d; i0 = s; }
        else if (d < d1) { d2 = d1; i2 = i1; d1 = d; i1 = s; }
        else if (d < d2) { d2 = d; i2 = s; }
    }
    float w0 = 1.0f / (d0 + 1e-8f), w1 = 1.0f / (d1 + 1e-8f), w2 = 1.0f / (d2 + 1e-8f);
    float ws = w0 + w1 + w2;
    w0 /= ws; w1 /= ws; w2 /= ws;
    float* o = out + (size_t)t * (C1 + C2);
    if (p1) {
        const float* s1 = p1 + (size_t)t * C1;
        for (int c = 0; c < C1; ++c) o[c] = s1[c];
    }
    const float* P = p2 + (size_t)b * S * C2;
    for (int c = 0; c < C2; ++c)
        o[C1 + c] = w0 * P[(size_t)i0 * C2 + c] + w1 * P[(size_t)i1 * C2 + c] + w2 * P[(size_t)i2 * C2 + c];
}

// ---------------- head2: (B,N,128) @ (128,2) + b -> (B,2,N) ----------------
__global__ void k_head2(const float* __restrict__ h, const float* __restrict__ W,
                        const float* __restrict__ bias, float* __restrict__ out, int B, int Nn) {
    int t = blockIdx.x * blockDim.x + threadIdx.x;
    if (t >= B * Nn) return;
    int b = t / Nn, n = t % Nn;
    const float* x = h + (size_t)t * 128;
    float a0 = bias[0], a1 = bias[1];
    for (int c = 0; c < 128; ++c) {
        float xv = x[c];
        a0 += xv * W[c * 2 + 0];
        a1 += xv * W[c * 2 + 1];
    }
    out[((size_t)b * 2 + 0) * Nn + n] = a0;
    out[((size_t)b * 2 + 1) * Nn + n] = a1;
}

// ---------------- transpose l4_points (B,16,512) -> (B,512,16) ----------------
__global__ void k_tr_l4(const float* __restrict__ p, float* __restrict__ o, int B, int S, int C) {
    int t = blockIdx.x * blockDim.x + threadIdx.x;
    if (t >= B * S * C) return;
    int b = t / (S * C);
    int r = t % (S * C);
    int s = r / C, c = r % C;
    o[((size_t)b * C + c) * S + s] = p[t];
}

struct Layer { const float *w, *b, *g, *bt; const _Float16* wp; int cin, cout, kp; };

extern "C" void kernel_launch(void* const* d_in, const int* in_sizes, int n_in,
                              void* d_out, int out_size, void* d_ws, size_t ws_size,
                              hipStream_t stream) {
    (void)in_sizes; (void)n_in; (void)out_size; (void)ws_size;
    const int B = 8, N0 = 8192;
    const float* xyz = (const float*)d_in[0];

    int p = 1;
    auto GL = [&](int cin, int cout) {
        Layer l;
        l.w  = (const float*)d_in[p];
        l.b  = (const float*)d_in[p + 1];
        l.g  = (const float*)d_in[p + 2];
        l.bt = (const float*)d_in[p + 3];
        l.wp = nullptr;
        l.cin = cin; l.cout = cout;
        l.kp = (cin + 31) & ~31;
        p += 4;
        return l;
    };
    Layer sa1[3] = { GL(9, 32),   GL(32, 32),   GL(32, 64) };
    Layer sa2[3] = { GL(67, 64),  GL(64, 64),   GL(64, 128) };
    Layer sa3[3] = { GL(131, 128),GL(128, 128), GL(128, 256) };
    Layer sa4[3] = { GL(259, 256),GL(256, 256), GL(256, 512) };
    Layer fp4[2] = { GL(768, 256),GL(256, 256) };
    Layer fp3[2] = { GL(384, 256),GL(256, 256) };
    Layer fp2[2] = { GL(320, 256),GL(256, 128) };
    Layer fp1[3] = { GL(128, 128),GL(128, 128), GL(128, 128) };
    Layer head1  = GL(128, 128);
    const float* h2w = (const float*)d_in[p];
    const float* h2b = (const float*)d_in[p + 1];

    // ---- workspace bump allocator ----
    char* wsp = (char*)d_ws;
    size_t off = 0;
    auto allocb = [&](size_t bytes) { void* r = (void*)(wsp + off); off += ((bytes + 255) & ~(size_t)255); return r; };
    auto allocf = [&](size_t n) { return (float*)allocb(n * 4); };
    auto alloci = [&](size_t n) { return (int*)allocb(n * 4); };
    auto alloch = [&](size_t n) { return (_Float16*)allocb(n * 2); };

    auto nb = [](long long n, int t) { return (unsigned)((n + t - 1) / t); };

    // ---- pack all weights into WMMA B-fragment layout (f16) ----
    Layer* allL[9] = { sa1, sa2, sa3, sa4, fp4, fp3, fp2, fp1, &head1 };
    int    nL[9]   = { 3, 3, 3, 3, 2, 2, 2, 3, 1 };
    for (int i = 0; i < 9; ++i) {
        for (int j = 0; j < nL[i]; ++j) {
            Layer& l = allL[i][j];
            size_t total = (size_t)(l.kp >> 5) * (l.cout >> 4) * 512;
            _Float16* wp = alloch(total);
            k_packW<<<nb(total, 256), 256, 0, stream>>>(l.w, wp, l.cin, l.kp, l.cout);
            l.wp = wp;
        }
    }

    float* l0_xyz = allocf((size_t)B * N0 * 3);
    float* lx[5]; int Sl[5] = { N0, 1024, 256, 64, 16 };
    lx[0] = l0_xyz;
    lx[1] = allocf((size_t)B * 1024 * 3);
    lx[2] = allocf((size_t)B * 256 * 3);
    lx[3] = allocf((size_t)B * 64 * 3);
    lx[4] = allocf((size_t)B * 16 * 3);
    float* l1p = allocf((size_t)B * 1024 * 64);
    float* l2p = allocf((size_t)B * 256 * 128);
    float* l3p = allocf((size_t)B * 64 * 256);
    float* l4p = allocf((size_t)B * 16 * 512);
    float* pl3n = allocf((size_t)B * 64 * 256);
    float* pl2n = allocf((size_t)B * 256 * 256);
    float* pl1n = allocf((size_t)B * 1024 * 128);
    int*   fidx = alloci((size_t)B * 1024);
    int*   gidx = alloci((size_t)B * 1024 * 32);
    float* arenaA = allocf((size_t)262144 * 64);  // largest: SA1 layer-3 output
    float* arenaB = allocf((size_t)262144 * 32);
    float* arenaC = allocf((size_t)262144 * 32);  // grouped / concat inputs (padded K)

    auto gemm = [&](const float* A, const Layer& L, float* O, int M) {
        dim3 grid((unsigned)(M / 64), (unsigned)(L.cout / 32));
        k_gemm<<<grid, 128, 0, stream>>>(A, L.wp, L.b, L.g, L.bt, O, M, L.kp, L.cout);
    };

    // ---- l0_xyz ----
    k_slice_xyz<<<nb(B * N0, 256), 256, 0, stream>>>(xyz, l0_xyz, B * N0);

    // ---- SA stages ----
    struct SAcfg { int lvl; int S; float r; const float* pts; int C; Layer* L; float* outp; };
    SAcfg sas[4] = {
        { 0, 1024, 0.1f, xyz, 6,  sa1, l1p },
        { 1, 256,  0.2f, l1p, 64, sa2, l2p },
        { 2, 64,   0.4f, l2p, 128, sa3, l3p },
        { 3, 16,   0.8f, l3p, 256, sa4, l4p },
    };
    for (int i = 0; i < 4; ++i) {
        const SAcfg& c = sas[i];
        int Np = Sl[c.lvl];
        const float* pxyz = lx[c.lvl];
        float* nxyz = lx[c.lvl + 1];
        k_fps<<<B, FPS_T, 0, stream>>>(pxyz, fidx, Np, c.S);
        k_gather_xyz<<<nb(B * c.S, 256), 256, 0, stream>>>(pxyz, fidx, nxyz, B, Np, c.S);
        k_ballquery<<<nb(B * c.S, 256), 256, 0, stream>>>(pxyz, nxyz, gidx, B, Np, c.S, c.r * c.r, 32);
        int M = B * c.S * 32;
        k_group<<<nb(M, 256), 256, 0, stream>>>(pxyz, c.pts, nxyz, gidx, arenaC, B, Np, c.S, 32, c.C, c.L[0].kp);
        gemm(arenaC, c.L[0], arenaA, M);
        gemm(arenaA, c.L[1], arenaB, M);
        gemm(arenaB, c.L[2], arenaA, M);
        k_maxpool<<<nb((long long)B * c.S * c.L[2].cout, 256), 256, 0, stream>>>(arenaA, c.outp, B * c.S, 32, c.L[2].cout);
    }

    // ---- FP stages ----
    { // fp4: l3 (N=64) <- l4 (S=16): concat(256, 512) -> 768 -> 256 -> 256 (pl3n)
        int Nn = 64, S = 16, M = B * Nn;
        k_fp_interp<<<nb(M, 256), 256, 0, stream>>>(lx[3], lx[4], l3p, 256, l4p, 512, arenaC, B, Nn, S);
        gemm(arenaC, fp4[0], arenaA, M);
        gemm(arenaA, fp4[1], pl3n, M);
    }
    { // fp3: l2 (N=256) <- l3 (S=64): concat(128, 256) -> 384 -> 256 -> 256 (pl2n)
        int Nn = 256, S = 64, M = B * Nn;
        k_fp_interp<<<nb(M, 256), 256, 0, stream>>>(lx[2], lx[3], l2p, 128, pl3n, 256, arenaC, B, Nn, S);
        gemm(arenaC, fp3[0], arenaA, M);
        gemm(arenaA, fp3[1], pl2n, M);
    }
    { // fp2: l1 (N=1024) <- l2 (S=256): concat(64, 256) -> 320 -> 256 -> 128 (pl1n)
        int Nn = 1024, S = 256, M = B * Nn;
        k_fp_interp<<<nb(M, 256), 256, 0, stream>>>(lx[1], lx[2], l1p, 64, pl2n, 256, arenaC, B, Nn, S);
        gemm(arenaC, fp2[0], arenaA, M);
        gemm(arenaA, fp2[1], pl1n, M);
    }
    { // fp1: l0 (N=8192) <- l1 (S=1024): interp(128) -> 128 -> 128 -> 128 (arenaC)
        int Nn = N0, S = 1024, M = B * Nn;
        k_fp_interp<<<nb(M, 256), 256, 0, stream>>>(lx[0], lx[1], (const float*)nullptr, 0, pl1n, 128, arenaC, B, Nn, S);
        gemm(arenaC, fp1[0], arenaA, M);
        gemm(arenaA, fp1[1], arenaB, M);
        gemm(arenaB, fp1[2], arenaC, M);
    }
    { // head1: 128 -> 128 (bn+relu); head2: linear 128 -> 2 -> (B,2,N); plus l4^T
        int M = B * N0;
        gemm(arenaC, head1, arenaA, M);
        float* out0 = (float*)d_out;
        k_head2<<<nb(M, 256), 256, 0, stream>>>(arenaA, h2w, h2b, out0, B, N0);
        float* out1 = out0 + (size_t)B * 2 * N0;
        k_tr_l4<<<nb(B * 16 * 512, 256), 256, 0, stream>>>(l4p, out1, B, 16, 512);
    }
}
